// LSTM_43404939493784
// MI455X (gfx1250) — compile-verified, matrix operand order checked by reading, not run on hardware
//
#include <hip/hip_runtime.h>

#define BATCH 131072
#define TT 15
#define HD 64
#define GG 256     // 4*H
#define OD 64
#define EPSV 1e-5f

typedef __attribute__((ext_vector_type(16))) _Float16 v16h;
typedef __attribute__((ext_vector_type(8)))  _Float16 v8h;
typedef __attribute__((ext_vector_type(8)))  float    v8f;

union AFrag { v16h v; v8h h[2]; };

// ---- fast transcendentals: CDNA5 has native v_tanh_f32 / v_exp_f32 / v_rcp_f32 ----
#if __has_builtin(__builtin_amdgcn_tanhf)
#define TANHF(x) __builtin_amdgcn_tanhf(x)
#define TANH_HW 1
#elif __has_builtin(__builtin_amdgcn_tanh_f32)
#define TANHF(x) __builtin_amdgcn_tanh_f32(x)
#define TANH_HW 1
#endif

__device__ __forceinline__ float sigm(float x){
#ifdef TANH_HW
  return __builtin_fmaf(0.5f, TANHF(0.5f*x), 0.5f);
#else
  return __builtin_amdgcn_rcpf(1.0f + __builtin_amdgcn_exp2f(-1.44269504f*x));
#endif
}
__device__ __forceinline__ float tanh_f(float x){
#ifdef TANH_HW
  return TANHF(x);
#else
  return 1.0f - 2.0f*__builtin_amdgcn_rcpf(1.0f + __builtin_amdgcn_exp2f(2.88539008f*x));
#endif
}

// Build a 16x32 f16 WMMA operand fragment from a row-major [rows][64] f16 LDS array.
// Per CDNA5 ISA layout: lanes 0-15 -> K runs [kb..kb+7],[kb+16..kb+23];
// lanes 16-31 -> [kb+8..kb+15],[kb+24..kb+31].  (B uses lane=N symmetric form.)
__device__ __forceinline__ v16h ld_frag(const _Float16* p0, int row, int kb, int half){
  AFrag f;
  const _Float16* p = p0 + row*HD + kb + (half<<3);
  f.h[0] = *(const v8h*)p;
  f.h[1] = *(const v8h*)(p + 16);
  return f.v;
}

// ---------------- BN stats: deterministic block partials ----------------
__global__ void bn_partial(const float* __restrict__ x, float* __restrict__ part){
  const int tid = threadIdx.x;
  const int gid = blockIdx.x*256 + tid;
  float s[TT], q[TT];
  #pragma unroll
  for(int t=0;t<TT;t++){ s[t]=0.f; q[t]=0.f; }
  for(long r=gid; r<BATCH; r+=(long)gridDim.x*256){
    const float* p = x + (size_t)r*TT;
    #pragma unroll
    for(int t=0;t<TT;t++){ float v=p[t]; s[t]+=v; q[t]+=v*v; }
  }
  __shared__ float red[8];
  float* outp = part + (size_t)blockIdx.x*32;
  #pragma unroll 1
  for(int t=0;t<TT;t++){
    float sv=s[t];
    #pragma unroll
    for(int m=16;m>=1;m>>=1) sv += __shfl_xor(sv, m, 32);
    __syncthreads();
    if((tid&31)==0) red[tid>>5]=sv;
    __syncthreads();
    if(tid==0){ float a=0.f; for(int wv=0;wv<8;wv++) a+=red[wv]; outp[t]=a; }
    float qv=q[t];
    #pragma unroll
    for(int m=16;m>=1;m>>=1) qv += __shfl_xor(qv, m, 32);
    __syncthreads();
    if((tid&31)==0) red[tid>>5]=qv;
    __syncthreads();
    if(tid==0){ float a=0.f; for(int wv=0;wv<8;wv++) a+=red[wv]; outp[16+t]=a; }
  }
}

__global__ void bn_finalize(float* __restrict__ ws, const float* __restrict__ gamma,
                            const float* __restrict__ beta, int nblk){
  const int t = threadIdx.x;
  if(t >= TT) return;
  const float* part = ws + 64;
  float s=0.f, q=0.f;
  for(int b=0;b<nblk;b++){ s += part[b*32+t]; q += part[b*32+16+t]; }
  float mean = s/(float)BATCH;
  float var  = q/(float)BATCH - mean*mean;
  float rs   = rsqrtf(var + EPSV);
  float sc   = rs*gamma[t];
  ws[t]      = sc;               // scale
  ws[16+t]   = beta[t]-mean*sc;  // shift
}

// ---------------- Fused BN-apply + 2-layer LSTM + FC ----------------
__global__ __launch_bounds__(256) void lstm_fused(
    const float* __restrict__ x,    const float* __restrict__ ws,
    const float* __restrict__ wih0, const float* __restrict__ whh0,
    const float* __restrict__ bih0, const float* __restrict__ bhh0,
    const float* __restrict__ wih1, const float* __restrict__ whh1,
    const float* __restrict__ bih1, const float* __restrict__ bhh1,
    const float* __restrict__ wfc,  const float* __restrict__ bfc,
    float* __restrict__ out)
{
  __shared__ __align__(16) _Float16 sWhh0[GG*HD];
  __shared__ __align__(16) _Float16 sWih1[GG*HD];
  __shared__ __align__(16) _Float16 sWhh1[GG*HD];
  __shared__ __align__(16) _Float16 sWfc[OD*HD];
  __shared__ __align__(16) float    sXn[128][16];
  __shared__ __align__(16) _Float16 sH0[2][8][16*HD];   // double-buffered per wave
  __shared__ __align__(16) _Float16 sH1[2][8][16*HD];
  __shared__ float sB0[GG], sB1[GG], sWi0[GG], sBfc[OD];

  const int tid = threadIdx.x;
  const int blk = blockIdx.x;

  for(int i=tid;i<GG*HD;i+=256){
    sWhh0[i]=(_Float16)whh0[i];
    sWih1[i]=(_Float16)wih1[i];
    sWhh1[i]=(_Float16)whh1[i];
  }
  for(int i=tid;i<OD*HD;i+=256) sWfc[i]=(_Float16)wfc[i];
  for(int i=tid;i<GG;i+=256){ sB0[i]=bih0[i]+bhh0[i]; sB1[i]=bih1[i]+bhh1[i]; sWi0[i]=wih0[i]; }
  if(tid<OD) sBfc[tid]=bfc[tid];
  for(int i=tid;i<128*TT;i+=256){
    int r=i/TT, t=i-r*TT;
    sXn[r][t] = x[(size_t)(blk*128+r)*TT + t]*ws[t] + ws[16+t];
  }
  for(int i=tid;i<2*8*16*HD;i+=256){
    (&sH0[0][0][0])[i]=(_Float16)0.0f;
    (&sH1[0][0][0])[i]=(_Float16)0.0f;
  }
  __syncthreads();

  const int lane=tid&31, w=tid>>5;
  const int cn=lane&15, half=lane>>4;

  float b0v[4][4], b1v[4][4], w0v[4][4];
  #pragma unroll
  for(int g=0;g<4;g++)
    #pragma unroll
    for(int q=0;q<4;q++){
      int col=q*64+g*16+cn;
      b0v[g][q]=sB0[col]; b1v[g][q]=sB1[col]; w0v[g][q]=sWi0[col];
    }

  v8f c0[4], c1[4];
  #pragma unroll
  for(int g=0;g<4;g++)
    #pragma unroll
    for(int e=0;e<8;e++){ c0[g][e]=0.f; c1[g][e]=0.f; }

  _Float16* h0b[2] = { &sH0[0][w][0], &sH0[1][w][0] };
  _Float16* h1b[2] = { &sH1[0][w][0], &sH1[1][w][0] };
  int cur=0;

  for(int t=0;t<TT;t++){
    const int nxt = cur^1;
    float xnv[8];
    #pragma unroll
    for(int e=0;e<8;e++) xnv[e]=sXn[w*16 + e + half*8][t];

    // ------- layer 0: acc = h0@Whh0^T (inline-0 C start); bias + xn*Wih0 deferred -------
    #pragma unroll
    for(int g=0;g<4;g++){
      v8f acc[4] = {};   // first WMMA takes inline 0 as C
      #pragma unroll
      for(int kb=0;kb<HD;kb+=32){
        v16h a = ld_frag(h0b[cur], cn, kb, half);
        #pragma unroll
        for(int q=0;q<4;q++){
          v16h b = ld_frag(sWhh0, q*64+g*16+cn, kb, half);
          acc[q]=__builtin_amdgcn_wmma_f32_16x16x32_f16(false,a,false,b,(short)0,acc[q],false,false);
        }
      }
      #pragma unroll
      for(int e=0;e<8;e++){
        float iv=sigm(acc[0][e] + __builtin_fmaf(xnv[e], w0v[g][0], b0v[g][0]));
        float fv=sigm(acc[1][e] + __builtin_fmaf(xnv[e], w0v[g][1], b0v[g][1]));
        float gv=tanh_f(acc[2][e] + __builtin_fmaf(xnv[e], w0v[g][2], b0v[g][2]));
        float ov=sigm(acc[3][e] + __builtin_fmaf(xnv[e], w0v[g][3], b0v[g][3]));
        float c = fv*c0[g][e] + iv*gv;
        c0[g][e]=c;
        float h = ov*tanh_f(c);
        h0b[nxt][(e+half*8)*HD + g*16+cn] = (_Float16)h;
      }
    }

    // ------- layer 1: acc = h0@Wih1^T + h1@Whh1^T (inline-0 C start); bias deferred -------
    #pragma unroll
    for(int g=0;g<4;g++){
      v8f acc[4] = {};
      #pragma unroll
      for(int kb=0;kb<HD;kb+=32){
        v16h a0 = ld_frag(h0b[nxt], cn, kb, half);
        v16h a1 = ld_frag(h1b[cur], cn, kb, half);
        #pragma unroll
        for(int q=0;q<4;q++){
          int n=q*64+g*16+cn;
          v16h bi = ld_frag(sWih1, n, kb, half);
          acc[q]=__builtin_amdgcn_wmma_f32_16x16x32_f16(false,a0,false,bi,(short)0,acc[q],false,false);
          v16h bh = ld_frag(sWhh1, n, kb, half);
          acc[q]=__builtin_amdgcn_wmma_f32_16x16x32_f16(false,a1,false,bh,(short)0,acc[q],false,false);
        }
      }
      #pragma unroll
      for(int e=0;e<8;e++){
        float iv=sigm(acc[0][e] + b1v[g][0]);
        float fv=sigm(acc[1][e] + b1v[g][1]);
        float gv=tanh_f(acc[2][e] + b1v[g][2]);
        float ov=sigm(acc[3][e] + b1v[g][3]);
        float c = fv*c1[g][e] + iv*gv;
        c1[g][e]=c;
        float h = ov*tanh_f(c);
        h1b[nxt][(e+half*8)*HD + g*16+cn] = (_Float16)h;
      }
    }
    cur = nxt;
  }

  // ---------------- FC tail: relu(h1_last @ Wfc^T + bfc) ----------------
  const size_t rowbase = (size_t)blk*128 + (size_t)w*16;
  #pragma unroll
  for(int ot=0;ot<4;ot++){
    v8f acc = {};
    #pragma unroll
    for(int kb=0;kb<HD;kb+=32){
      v16h a = ld_frag(h1b[cur], cn, kb, half);
      v16h b = ld_frag(sWfc, ot*16+cn, kb, half);
      acc=__builtin_amdgcn_wmma_f32_16x16x32_f16(false,a,false,b,(short)0,acc,false,false);
    }
    const float bb = sBfc[ot*16+cn];
    #pragma unroll
    for(int e=0;e<8;e++){
      float v=acc[e] + bb; v = v>0.f ? v : 0.f;
      __builtin_nontemporal_store(v, &out[(rowbase + e + half*8)*OD + ot*16 + cn]);
    }
  }
}

extern "C" void kernel_launch(void* const* d_in, const int* in_sizes, int n_in,
                              void* d_out, int out_size, void* d_ws, size_t ws_size,
                              hipStream_t stream){
  const float* x    =(const float*)d_in[0];
  const float* gamma=(const float*)d_in[1];
  const float* beta =(const float*)d_in[2];
  const float* wih0 =(const float*)d_in[3];
  const float* whh0 =(const float*)d_in[4];
  const float* bih0 =(const float*)d_in[5];
  const float* bhh0 =(const float*)d_in[6];
  const float* wih1 =(const float*)d_in[7];
  const float* whh1 =(const float*)d_in[8];
  const float* bih1 =(const float*)d_in[9];
  const float* bhh1 =(const float*)d_in[10];
  const float* wfc  =(const float*)d_in[11];
  const float* bfc  =(const float*)d_in[12];
  float* ws  = (float*)d_ws;
  float* out = (float*)d_out;

  const int NB = 256;
  bn_partial<<<NB, 256, 0, stream>>>(x, ws + 64);
  bn_finalize<<<1, 32, 0, stream>>>(ws, gamma, beta, NB);
  lstm_fused<<<BATCH/128, 256, 0, stream>>>(x, ws, wih0, whh0, bih0, bhh0,
                                            wih1, whh1, bih1, bhh1, wfc, bfc, out);
}
